// TspformerSearch_49950469653428
// MI455X (gfx1250) — matrix operand order repaired and checked by reading, not compiled, not run on hardware
//
#include <hip/hip_runtime.h>
#include <hip/hip_bf16.h>

// ---------------------------------------------------------------------------
// Tspformer search on gfx1250: WMMA-based GEMMs + VALU attention/normalization
// ---------------------------------------------------------------------------

#define BSZ   128
#define NN    50
#define SEQ   51          // NN + 1 (start placeholder)
#define DM    128
#define HH    8
#define DH    16
#define FFD   512
#define LE    6
#define LD    2
#define BS    (BSZ * SEQ) // 6528 rows, multiple of 16

typedef __attribute__((ext_vector_type(16))) _Float16 v16h;
typedef __attribute__((ext_vector_type(8)))  float    v8f;

// ---------------------------------------------------------------------------
// Weight pre-pass: transpose + convert to f16.  Wt[n][k] = (f16)W[k][n]
// ---------------------------------------------------------------------------
__global__ void convw_kernel(const float* __restrict__ W,
                             _Float16* __restrict__ Wt, int K, int N)
{
  int i = blockIdx.x * 256 + threadIdx.x;
  if (i >= K * N) return;
  int k = i / N, n = i - k * N;
  Wt[(size_t)n * K + k] = (_Float16)W[i];
}

// ---------------------------------------------------------------------------
// GEMM: C[M,N] = act(A[M,K] @ W[K,N] + bias) (+ resid)
// One wave per 16x64 output tile: one A fragment shared across FOUR
// v_wmma_f32_16x16x32_f16 per k-step (amortizes f32 A loads + f16 converts).
// A: f32 b128 loads.  W: pre-transposed f16, contiguous b128 loads.
// Compile-time RELU/RESID -> branch-free epilogue.
// ---------------------------------------------------------------------------
template <int K, int RELU, int RESID>
__global__ __launch_bounds__(32) void gemm_wmma_kernel(
    const float* __restrict__ A, int lda,
    const _Float16* __restrict__ Wt,     // [N][K]
    const float* __restrict__ bias,
    const float* __restrict__ resid,
    float* __restrict__ C, int ldc)
{
  const int lane = threadIdx.x;   // 0..31, all active (EXEC all-ones)
  const int half = lane >> 4;     // 0/1
  const int l16  = lane & 15;
  const int m0 = blockIdx.x * 16;
  const int n0 = blockIdx.y * 64;

  v8f c0, c1, c2, c3;
  float bi0 = bias ? bias[n0 + l16]      : 0.0f;
  float bi1 = bias ? bias[n0 + 16 + l16] : 0.0f;
  float bi2 = bias ? bias[n0 + 32 + l16] : 0.0f;
  float bi3 = bias ? bias[n0 + 48 + l16] : 0.0f;
#pragma unroll
  for (int r = 0; r < 8; ++r) { c0[r] = bi0; c1[r] = bi1; c2[r] = bi2; c3[r] = bi3; }

  const float*    Arow = A  + (size_t)(m0 + l16) * lda;
  const _Float16* W0   = Wt + (size_t)(n0 + l16) * K;
  const _Float16* W1   = Wt + (size_t)(n0 + 16 + l16) * K;
  const _Float16* W2   = Wt + (size_t)(n0 + 32 + l16) * K;
  const _Float16* W3   = Wt + (size_t)(n0 + 48 + l16) * K;

#pragma unroll 4
  for (int k0 = 0; k0 < K; k0 += 32) {
    // A fragment: a[j] = A[m][k0 + half*16 + j]
    float4 a0 = *(const float4*)(Arow + k0 + half * 16);
    float4 a1 = *(const float4*)(Arow + k0 + half * 16 + 4);
    float4 a2 = *(const float4*)(Arow + k0 + half * 16 + 8);
    float4 a3 = *(const float4*)(Arow + k0 + half * 16 + 12);
    v16h a;
    a[0]=(_Float16)a0.x;  a[1]=(_Float16)a0.y;  a[2]=(_Float16)a0.z;  a[3]=(_Float16)a0.w;
    a[4]=(_Float16)a1.x;  a[5]=(_Float16)a1.y;  a[6]=(_Float16)a1.z;  a[7]=(_Float16)a1.w;
    a[8]=(_Float16)a2.x;  a[9]=(_Float16)a2.y;  a[10]=(_Float16)a2.z; a[11]=(_Float16)a2.w;
    a[12]=(_Float16)a3.x; a[13]=(_Float16)a3.y; a[14]=(_Float16)a3.z; a[15]=(_Float16)a3.w;
    // B fragments: b[j] = W[k0 + half*16 + j][n]  (contiguous in Wt)
    v16h b0 = *(const v16h*)(W0 + k0 + half * 16);
    v16h b1 = *(const v16h*)(W1 + k0 + half * 16);
    v16h b2 = *(const v16h*)(W2 + k0 + half * 16);
    v16h b3 = *(const v16h*)(W3 + k0 + half * 16);
    c0 = __builtin_amdgcn_wmma_f32_16x16x32_f16(false, a, false, b0, (short)0, c0, false, false);
    c1 = __builtin_amdgcn_wmma_f32_16x16x32_f16(false, a, false, b1, (short)0, c1, false, false);
    c2 = __builtin_amdgcn_wmma_f32_16x16x32_f16(false, a, false, b2, (short)0, c2, false, false);
    c3 = __builtin_amdgcn_wmma_f32_16x16x32_f16(false, a, false, b3, (short)0, c3, false, false);
  }

#pragma unroll
  for (int r = 0; r < 8; ++r) {
    int row = m0 + (half << 3) + r;
    float v0 = c0[r], v1 = c1[r], v2 = c2[r], v3 = c3[r];
    if (RELU) {
      v0 = fmaxf(v0, 0.0f); v1 = fmaxf(v1, 0.0f);
      v2 = fmaxf(v2, 0.0f); v3 = fmaxf(v3, 0.0f);
    }
    if (RESID) {
      const float* rp = resid + (size_t)row * ldc + n0 + l16;
      v0 += rp[0]; v1 += rp[16]; v2 += rp[32]; v3 += rp[48];
    }
    float* cp = C + (size_t)row * ldc + n0 + l16;
    cp[0] = v0; cp[16] = v1; cp[32] = v2; cp[48] = v3;
  }
}

// ---------------------------------------------------------------------------
// Positional-encoding table pe[t][d], t<NN
// ---------------------------------------------------------------------------
__global__ void pe_kernel(float* __restrict__ pe)
{
  int t = blockIdx.x, d = threadIdx.x;
  float angle = (float)t / __powf(10000.0f, (float)(2 * (d / 2)) / (float)DM);
  pe[t * DM + d] = (d & 1) ? __cosf(angle) : __sinf(angle);
}

// ---------------------------------------------------------------------------
// Input embedding + start placeholder: h[B,SEQ,D]
// ---------------------------------------------------------------------------
__global__ void embed_kernel(const float* __restrict__ x,
                             const float* __restrict__ w_in,
                             const float* __restrict__ b_in,
                             const float* __restrict__ start_ph,
                             float* __restrict__ h)
{
  int bs = blockIdx.x;          // 0..BS-1
  int d  = threadIdx.x;         // 0..127
  int b = bs / SEQ, s = bs % SEQ;
  float v;
  if (s < NN) {
    const float* xp = x + (size_t)(b * NN + s) * 2;
    v = xp[0] * w_in[d] + xp[1] * w_in[DM + d] + b_in[d];
  } else {
    v = start_ph[d];
  }
  h[(size_t)bs * DM + d] = v;
}

// ---------------------------------------------------------------------------
// Encoder MHA core: flash-style softmax, one thread per (b,h,query)
// ---------------------------------------------------------------------------
__global__ void enc_attn_kernel(const float* __restrict__ Q,
                                const float* __restrict__ Kb,
                                const float* __restrict__ Vb,
                                float* __restrict__ O)
{
  int bh = blockIdx.x;          // B*H blocks
  int b = bh >> 3, h = bh & 7;
  int qs = threadIdx.x;         // 64 threads, 51 active
  if (qs >= SEQ) return;

  float q[DH], acc[DH];
  const float* qp = Q + ((size_t)(b * SEQ + qs) * DM) + h * DH;
#pragma unroll
  for (int d = 0; d < DH; ++d) { q[d] = qp[d]; acc[d] = 0.0f; }
  float m = -1e30f, l = 0.0f;

  for (int ks = 0; ks < SEQ; ++ks) {
    const float* kp = Kb + ((size_t)(b * SEQ + ks) * DM) + h * DH;
    float s = 0.0f;
#pragma unroll
    for (int d = 0; d < DH; ++d) s += q[d] * kp[d];
    s *= 0.25f;                               // 1/sqrt(16)
    float mn = fmaxf(m, s);
    float sc = __expf(m - mn);
    float p  = __expf(s - mn);
    l = l * sc + p;
    const float* vp = Vb + ((size_t)(b * SEQ + ks) * DM) + h * DH;
#pragma unroll
    for (int d = 0; d < DH; ++d) acc[d] = acc[d] * sc + p * vp[d];
    m = mn;
  }
  float inv = 1.0f / l;
  float* op = O + ((size_t)(b * SEQ + qs) * DM) + h * DH;
#pragma unroll
  for (int d = 0; d < DH; ++d) op[d] = acc[d] * inv;
}

// ---------------------------------------------------------------------------
// BatchNorm over (batch, nodes) per feature, in-place. One block per feature.
// ---------------------------------------------------------------------------
__global__ void bn_kernel(float* __restrict__ X,
                          const float* __restrict__ gamma,
                          const float* __restrict__ beta, int M)
{
  int d = blockIdx.x, t = threadIdx.x;  // 256 threads
  __shared__ float ssum[256], ssq[256];
  float s = 0.0f, sq = 0.0f;
  for (int r = t; r < M; r += 256) {
    float v = X[(size_t)r * DM + d];
    s += v; sq += v * v;
  }
  ssum[t] = s; ssq[t] = sq; __syncthreads();
  for (int off = 128; off > 0; off >>= 1) {
    if (t < off) { ssum[t] += ssum[t + off]; ssq[t] += ssq[t + off]; }
    __syncthreads();
  }
  float mean = ssum[0] / (float)M;
  float var  = ssq[0] / (float)M - mean * mean;
  float g = gamma[d], be = beta[d];
  float rs = rsqrtf(var + 1e-5f);
  for (int r = t; r < M; r += 256) {
    size_t i = (size_t)r * DM + d;
    X[i] = g * (X[i] - mean) * rs + be;
  }
}

// ---------------------------------------------------------------------------
// LayerNorm over last dim (128), in-place. One block (128 threads) per row.
// ---------------------------------------------------------------------------
__global__ void ln_kernel(float* __restrict__ X,
                          const float* __restrict__ gamma,
                          const float* __restrict__ beta)
{
  int b = blockIdx.x, d = threadIdx.x;
  __shared__ float sh[DM];
  size_t i = (size_t)b * DM + d;
  float x = X[i];
  sh[d] = x; __syncthreads();
  for (int off = 64; off > 0; off >>= 1) { if (d < off) sh[d] += sh[d + off]; __syncthreads(); }
  float mean = sh[0] / (float)DM;
  __syncthreads();
  float dv = x - mean;
  sh[d] = dv * dv; __syncthreads();
  for (int off = 64; off > 0; off >>= 1) { if (d < off) sh[d] += sh[d + off]; __syncthreads(); }
  float var = sh[0] / (float)DM;
  X[i] = gamma[d] * dv * rsqrtf(var + 1e-5f) + beta[d];
}

// ---------------------------------------------------------------------------
// Decoder: gather current token h_t = h_enc[b, idx[b]] + pe[t]
// ---------------------------------------------------------------------------
__global__ void dec_gather_kernel(const float* __restrict__ h_enc,
                                  const int* __restrict__ idx,
                                  const float* __restrict__ pe,
                                  int t, float* __restrict__ ht)
{
  int b = blockIdx.x, d = threadIdx.x;
  ht[(size_t)b * DM + d] =
      h_enc[((size_t)b * SEQ + idx[b]) * DM + d] + pe[t * DM + d];
}

// ---------------------------------------------------------------------------
// Decoder self-attention with KV-cache, one thread per (b, h)
// ---------------------------------------------------------------------------
__global__ void dec_self_attn_kernel(const float* __restrict__ qb,
                                     const float* __restrict__ kb,
                                     const float* __restrict__ vb,
                                     float* __restrict__ Kc,
                                     float* __restrict__ Vc,
                                     float* __restrict__ O, int l, int t)
{
  int tid = blockIdx.x * blockDim.x + threadIdx.x;  // 1024 threads
  if (tid >= BSZ * HH) return;
  int b = tid >> 3, h = tid & 7;

  // write this step's k/v into cache [LD][B][H][NN][DH]
  size_t cbase = ((((size_t)l * BSZ + b) * HH + h) * NN + t) * DH;
  const float* kp = kb + (size_t)b * DM + h * DH;
  const float* vp = vb + (size_t)b * DM + h * DH;
#pragma unroll
  for (int d = 0; d < DH; ++d) { Kc[cbase + d] = kp[d]; Vc[cbase + d] = vp[d]; }

  float q[DH], acc[DH];
  const float* qp = qb + (size_t)b * DM + h * DH;
#pragma unroll
  for (int d = 0; d < DH; ++d) { q[d] = qp[d]; acc[d] = 0.0f; }
  float m = -1e30f, l_ = 0.0f;

  for (int s = 0; s <= t; ++s) {
    size_t kb2 = ((((size_t)l * BSZ + b) * HH + h) * NN + s) * DH;
    float sc = 0.0f;
#pragma unroll
    for (int d = 0; d < DH; ++d) sc += q[d] * Kc[kb2 + d];
    sc *= 0.25f;
    float mn = fmaxf(m, sc);
    float scale = __expf(m - mn);
    float p = __expf(sc - mn);
    l_ = l_ * scale + p;
#pragma unroll
    for (int d = 0; d < DH; ++d) acc[d] = acc[d] * scale + p * Vc[kb2 + d];
    m = mn;
  }
  float inv = 1.0f / l_;
  float* op = O + (size_t)b * DM + h * DH;
#pragma unroll
  for (int d = 0; d < DH; ++d) op[d] = acc[d] * inv;
}

// ---------------------------------------------------------------------------
// Decoder cross-attention over precomputed K_att/V_att [B,SEQ,LD,H,DH]
// ---------------------------------------------------------------------------
__global__ void dec_cross_attn_kernel(const float* __restrict__ qb,
                                      const float* __restrict__ K_att,
                                      const float* __restrict__ V_att,
                                      float* __restrict__ O, int l)
{
  int tid = blockIdx.x * blockDim.x + threadIdx.x;
  if (tid >= BSZ * HH) return;
  int b = tid >> 3, h = tid & 7;

  float q[DH], acc[DH];
  const float* qp = qb + (size_t)b * DM + h * DH;
#pragma unroll
  for (int d = 0; d < DH; ++d) { q[d] = qp[d]; acc[d] = 0.0f; }
  float m = -1e30f, l_ = 0.0f;

  for (int n = 0; n < SEQ; ++n) {
    size_t base = (((size_t)(b * SEQ + n) * LD + l) * HH + h) * DH;
    float sc = 0.0f;
#pragma unroll
    for (int d = 0; d < DH; ++d) sc += q[d] * K_att[base + d];
    sc *= 0.25f;
    float mn = fmaxf(m, sc);
    float scale = __expf(m - mn);
    float p = __expf(sc - mn);
    l_ = l_ * scale + p;
#pragma unroll
    for (int d = 0; d < DH; ++d) acc[d] = acc[d] * scale + p * V_att[base + d];
    m = mn;
  }
  float inv = 1.0f / l_;
  float* op = O + (size_t)b * DM + h * DH;
#pragma unroll
  for (int d = 0; d < DH; ++d) op[d] = acc[d] * inv;
}

// ---------------------------------------------------------------------------
// Decoder state init
// ---------------------------------------------------------------------------
__global__ void dec_init_kernel(int* __restrict__ mask, int* __restrict__ idx,
                                float* __restrict__ slp)
{
  int b = blockIdx.x, n = threadIdx.x;  // 64 threads
  if (n < SEQ) mask[b * SEQ + n] = (n == NN) ? 1 : 0;
  if (n == 0) { idx[b] = NN; slp[b] = 0.0f; }
}

// ---------------------------------------------------------------------------
// Pointer head: logits -> masked log-softmax -> argmax -> state update
// ---------------------------------------------------------------------------
__global__ void pointer_kernel(const float* __restrict__ qf,
                               const float* __restrict__ K_fin,
                               int* __restrict__ mask, int* __restrict__ idx,
                               float* __restrict__ slp,
                               float* __restrict__ out_tours,
                               float* __restrict__ out_scores,
                               int t, int writeScores)
{
  int b = blockIdx.x, n = threadIdx.x;  // 64 threads, 51 active
  __shared__ float sval[64];
  __shared__ int   sidx[64];

  float lg = -3.0e38f;
  if (n < SEQ) {
    const float* qp = qf + (size_t)b * DM;
    const float* kp = K_fin + (size_t)(b * SEQ + n) * DM;
    float s = 0.0f;
    for (int d = 0; d < DM; ++d) s += qp[d] * kp[d];
    s *= 0.0883883476483f;                 // 1/sqrt(128)
    lg = 10.0f * tanhf(s);
    if (mask[b * SEQ + n]) lg = -1e9f;
  }

  // argmax (smallest index on ties, matching jnp.argmax)
  sval[n] = lg; sidx[n] = n; __syncthreads();
  for (int off = 32; off > 0; off >>= 1) {
    if (n < off) {
      float vo = sval[n + off]; int io = sidx[n + off];
      if (vo > sval[n] || (vo == sval[n] && io < sidx[n])) { sval[n] = vo; sidx[n] = io; }
    }
    __syncthreads();
  }
  float mx = sval[0];
  int nxt = sidx[0];
  __syncthreads();

  float e = (n < SEQ) ? __expf(lg - mx) : 0.0f;
  sval[n] = e; __syncthreads();
  for (int off = 32; off > 0; off >>= 1) {
    if (n < off) sval[n] += sval[n + off];
    __syncthreads();
  }

  if (n == 0) {
    // logp[nxt] = logits[nxt] - mx - log(sum) = -log(sum)  (nxt is argmax)
    float ns = slp[b] - __logf(sval[0]);
    slp[b] = ns;
    mask[b * SEQ + nxt] = 1;
    idx[b] = nxt;
    out_tours[(size_t)b * NN + t] = (float)nxt;
    if (writeScores) out_scores[b] = ns;
  }
}

// ---------------------------------------------------------------------------
// Host-side orchestration
// ---------------------------------------------------------------------------
extern "C" void kernel_launch(void* const* d_in, const int* in_sizes, int n_in,
                              void* d_out, int out_size, void* d_ws, size_t ws_size,
                              hipStream_t stream)
{
  const float* x          = (const float*)d_in[0];
  const float* w_input    = (const float*)d_in[1];
  const float* b_input    = (const float*)d_in[2];
  const float* start_ph   = (const float*)d_in[3];
  const float* enc_attn_w = (const float*)d_in[4];   // [LE,4,D,D]
  const float* enc_attn_b = (const float*)d_in[5];   // [LE,4,D]
  const float* enc_ffn_w1 = (const float*)d_in[6];   // [LE,D,FF]
  const float* enc_ffn_b1 = (const float*)d_in[7];
  const float* enc_ffn_w2 = (const float*)d_in[8];   // [LE,FF,D]
  const float* enc_ffn_b2 = (const float*)d_in[9];
  const float* enc_bn     = (const float*)d_in[10];  // [LE,2,2,D]
  const float* dec_self_w = (const float*)d_in[11];  // [LD,4,D,D]
  const float* dec_self_b = (const float*)d_in[12];
  const float* dec_cross_w= (const float*)d_in[13];  // [LD,2,D,D]
  const float* dec_cross_b= (const float*)d_in[14];
  const float* dec_ffn_w1 = (const float*)d_in[15];  // [LD,D,FF]
  const float* dec_ffn_b1 = (const float*)d_in[16];
  const float* dec_ffn_w2 = (const float*)d_in[17];  // [LD,FF,D]
  const float* dec_ffn_b2 = (const float*)d_in[18];
  const float* dec_ln     = (const float*)d_in[19];  // [LD,3,2,D]
  const float* wk_dec     = (const float*)d_in[20];  // [D, LD*D]
  const float* bk_dec     = (const float*)d_in[21];
  const float* wv_dec     = (const float*)d_in[22];
  const float* bv_dec     = (const float*)d_in[23];
  const float* wq_final   = (const float*)d_in[24];  // [D,D]
  const float* wk_final   = (const float*)d_in[25];  // [D,D]
  (void)in_sizes; (void)n_in; (void)out_size; (void)ws_size;

  // ---- workspace carve-up (256B aligned) ----
  size_t cur = 0;
  auto alloc = [&](size_t nElems, size_t esz) -> void* {
    cur = (cur + 255) & ~(size_t)255;
    void* p = (char*)d_ws + cur;
    cur += nElems * esz;
    return p;
  };
  auto allocf = [&](size_t n) { return (float*)alloc(n, 4); };
  auto alloch = [&](size_t n) { return (_Float16*)alloc(n, 2); };

  float* h0    = allocf((size_t)BS * DM);
  float* h1    = allocf((size_t)BS * DM);
  float* qb    = allocf((size_t)BS * DM);
  float* kb    = allocf((size_t)BS * DM);
  float* vb    = allocf((size_t)BS * DM);
  float* ab    = allocf((size_t)BS * DM);
  float* ffb   = allocf((size_t)BS * FFD);
  float* K_att = allocf((size_t)BS * LD * DM);
  float* V_att = allocf((size_t)BS * LD * DM);
  float* K_fin = allocf((size_t)BS * DM);
  float* pe    = allocf((size_t)NN * DM);
  float* ht0   = allocf((size_t)BSZ * DM);
  float* ht1   = allocf((size_t)BSZ * DM);
  float* qt    = allocf((size_t)BSZ * DM);
  float* kt    = allocf((size_t)BSZ * DM);
  float* vt    = allocf((size_t)BSZ * DM);
  float* at    = allocf((size_t)BSZ * DM);
  float* fft   = allocf((size_t)BSZ * FFD);
  float* Kc    = allocf((size_t)LD * BSZ * HH * NN * DH);
  float* Vc    = allocf((size_t)LD * BSZ * HH * NN * DH);
  int*   maskp = (int*)alloc((size_t)BSZ * SEQ, 4);
  int*   idxp  = (int*)alloc((size_t)BSZ, 4);
  float* slp   = allocf((size_t)BSZ);

  // f16 transposed weights
  _Float16* encAWt = alloch((size_t)LE * 4 * DM * DM);
  _Float16* encF1t = alloch((size_t)LE * DM * FFD);
  _Float16* encF2t = alloch((size_t)LE * FFD * DM);
  _Float16* decSWt = alloch((size_t)LD * 4 * DM * DM);
  _Float16* decCWt = alloch((size_t)LD * 2 * DM * DM);
  _Float16* decF1t = alloch((size_t)LD * DM * FFD);
  _Float16* decF2t = alloch((size_t)LD * FFD * DM);
  _Float16* wkDt   = alloch((size_t)DM * LD * DM);
  _Float16* wvDt   = alloch((size_t)DM * LD * DM);
  _Float16* wqFt   = alloch((size_t)DM * DM);
  _Float16* wkFt   = alloch((size_t)DM * DM);

  float* out_tours  = (float*)d_out;            // [B, NN] as float
  float* out_scores = (float*)d_out + BSZ * NN; // [B]

  auto convw = [&](const float* W, _Float16* Wt, int K, int N) {
    convw_kernel<<<(K * N + 255) / 256, 256, 0, stream>>>(W, Wt, K, N);
  };
  // K=128, plain / +resid / +relu ; K=512, +resid.  Grid: (M/16, N/64).
  auto g128   = [&](const float* A, const _Float16* Wt, const float* bias,
                    float* C, int ldc, int M, int N) {
    gemm_wmma_kernel<128, 0, 0><<<dim3(M / 16, N / 64), 32, 0, stream>>>(A, 128, Wt, bias, nullptr, C, ldc);
  };
  auto g128r  = [&](const float* A, const _Float16* Wt, const float* bias,
                    const float* resid, float* C, int ldc, int M, int N) {
    gemm_wmma_kernel<128, 0, 1><<<dim3(M / 16, N / 64), 32, 0, stream>>>(A, 128, Wt, bias, resid, C, ldc);
  };
  auto g128relu = [&](const float* A, const _Float16* Wt, const float* bias,
                      float* C, int ldc, int M, int N) {
    gemm_wmma_kernel<128, 1, 0><<<dim3(M / 16, N / 64), 32, 0, stream>>>(A, 128, Wt, bias, nullptr, C, ldc);
  };
  auto g512r  = [&](const float* A, const _Float16* Wt, const float* bias,
                    const float* resid, float* C, int ldc, int M, int N) {
    gemm_wmma_kernel<512, 0, 1><<<dim3(M / 16, N / 64), 32, 0, stream>>>(A, 512, Wt, bias, resid, C, ldc);
  };

  // ---- one-time weight transpose/convert ----
  for (int i = 0; i < LE * 4; ++i)
    convw(enc_attn_w + (size_t)i * DM * DM, encAWt + (size_t)i * DM * DM, DM, DM);
  for (int l = 0; l < LE; ++l) {
    convw(enc_ffn_w1 + (size_t)l * DM * FFD, encF1t + (size_t)l * DM * FFD, DM, FFD);
    convw(enc_ffn_w2 + (size_t)l * FFD * DM, encF2t + (size_t)l * FFD * DM, FFD, DM);
  }
  for (int i = 0; i < LD * 4; ++i)
    convw(dec_self_w + (size_t)i * DM * DM, decSWt + (size_t)i * DM * DM, DM, DM);
  for (int i = 0; i < LD * 2; ++i)
    convw(dec_cross_w + (size_t)i * DM * DM, decCWt + (size_t)i * DM * DM, DM, DM);
  for (int l = 0; l < LD; ++l) {
    convw(dec_ffn_w1 + (size_t)l * DM * FFD, decF1t + (size_t)l * DM * FFD, DM, FFD);
    convw(dec_ffn_w2 + (size_t)l * FFD * DM, decF2t + (size_t)l * FFD * DM, FFD, DM);
  }
  convw(wk_dec, wkDt, DM, LD * DM);
  convw(wv_dec, wvDt, DM, LD * DM);
  convw(wq_final, wqFt, DM, DM);
  convw(wk_final, wkFt, DM, DM);

  // ---- static tables + embedding ----
  pe_kernel<<<NN, DM, 0, stream>>>(pe);
  embed_kernel<<<BS, DM, 0, stream>>>(x, w_input, b_input, start_ph, h0);

  // ---- encoder ----
  float* hcur = h0;
  float* hnext = h1;
  for (int l = 0; l < LE; ++l) {
    const _Float16* wq = encAWt + (size_t)(l * 4 + 0) * DM * DM;
    const _Float16* wk = encAWt + (size_t)(l * 4 + 1) * DM * DM;
    const _Float16* wv = encAWt + (size_t)(l * 4 + 2) * DM * DM;
    const _Float16* wo = encAWt + (size_t)(l * 4 + 3) * DM * DM;
    const float* bq = enc_attn_b + (size_t)(l * 4 + 0) * DM;
    const float* bk = enc_attn_b + (size_t)(l * 4 + 1) * DM;
    const float* bv = enc_attn_b + (size_t)(l * 4 + 2) * DM;
    const float* bo = enc_attn_b + (size_t)(l * 4 + 3) * DM;

    g128(hcur, wq, bq, qb, DM, BS, DM);
    g128(hcur, wk, bk, kb, DM, BS, DM);
    g128(hcur, wv, bv, vb, DM, BS, DM);
    enc_attn_kernel<<<BSZ * HH, 64, 0, stream>>>(qb, kb, vb, ab);
    g128r(ab, wo, bo, hcur, hnext, DM, BS, DM);
    {
      const float* g = enc_bn + (size_t)((l * 2 + 0) * 2 + 0) * DM;
      bn_kernel<<<DM, 256, 0, stream>>>(hnext, g, g + DM, BS);
    }
    float* tmp = hcur; hcur = hnext; hnext = tmp;

    g128relu(hcur, encF1t + (size_t)l * DM * FFD, enc_ffn_b1 + (size_t)l * FFD, ffb, FFD, BS, FFD);
    g512r(ffb, encF2t + (size_t)l * FFD * DM, enc_ffn_b2 + (size_t)l * DM, hcur, hnext, DM, BS, DM);
    {
      const float* g = enc_bn + (size_t)((l * 2 + 1) * 2 + 0) * DM;
      bn_kernel<<<DM, 256, 0, stream>>>(hnext, g, g + DM, BS);
    }
    tmp = hcur; hcur = hnext; hnext = tmp;
  }
  const float* h_enc = hcur;

  // ---- decoder precompute: cross K/V and final pointer keys ----
  g128(h_enc, wkDt, bk_dec, K_att, LD * DM, BS, LD * DM);
  g128(h_enc, wvDt, bv_dec, V_att, LD * DM, BS, LD * DM);
  g128(h_enc, wkFt, nullptr, K_fin, DM, BS, DM);

  dec_init_kernel<<<BSZ, 64, 0, stream>>>(maskp, idxp, slp);

  // ---- autoregressive decode ----
  float* hA = ht0;
  float* hB = ht1;
  for (int t = 0; t < NN; ++t) {
    dec_gather_kernel<<<BSZ, DM, 0, stream>>>(h_enc, idxp, pe, t, hA);

    for (int l = 0; l < LD; ++l) {
      const _Float16* wq = decSWt + (size_t)(l * 4 + 0) * DM * DM;
      const _Float16* wk = decSWt + (size_t)(l * 4 + 1) * DM * DM;
      const _Float16* wv = decSWt + (size_t)(l * 4 + 2) * DM * DM;
      const _Float16* wo = decSWt + (size_t)(l * 4 + 3) * DM * DM;
      const float* bq = dec_self_b + (size_t)(l * 4 + 0) * DM;
      const float* bk = dec_self_b + (size_t)(l * 4 + 1) * DM;
      const float* bv = dec_self_b + (size_t)(l * 4 + 2) * DM;
      const float* bo = dec_self_b + (size_t)(l * 4 + 3) * DM;

      g128(hA, wq, bq, qt, DM, BSZ, DM);
      g128(hA, wk, bk, kt, DM, BSZ, DM);
      g128(hA, wv, bv, vt, DM, BSZ, DM);
      dec_self_attn_kernel<<<8, 128, 0, stream>>>(qt, kt, vt, Kc, Vc, at, l, t);
      g128r(at, wo, bo, hA, hB, DM, BSZ, DM);
      {
        const float* g = dec_ln + (size_t)((l * 3 + 0) * 2 + 0) * DM;
        ln_kernel<<<BSZ, DM, 0, stream>>>(hB, g, g + DM);
      }
      { float* tmp = hA; hA = hB; hB = tmp; }

      const _Float16* cw0 = decCWt + (size_t)(l * 2 + 0) * DM * DM;
      const _Float16* cw1 = decCWt + (size_t)(l * 2 + 1) * DM * DM;
      const float* cb0 = dec_cross_b + (size_t)(l * 2 + 0) * DM;
      const float* cb1 = dec_cross_b + (size_t)(l * 2 + 1) * DM;

      g128(hA, cw0, cb0, qt, DM, BSZ, DM);
      dec_cross_attn_kernel<<<8, 128, 0, stream>>>(qt, K_att, V_att, at, l);
      g128r(at, cw1, cb1, hA, hB, DM, BSZ, DM);
      {
        const float* g = dec_ln + (size_t)((l * 3 + 1) * 2 + 0) * DM;
        ln_kernel<<<BSZ, DM, 0, stream>>>(hB, g, g + DM);
      }
      { float* tmp = hA; hA = hB; hB = tmp; }

      g128relu(hA, decF1t + (size_t)l * DM * FFD, dec_ffn_b1 + (size_t)l * FFD, fft, FFD, BSZ, FFD);
      g512r(fft, decF2t + (size_t)l * FFD * DM, dec_ffn_b2 + (size_t)l * DM, hA, hB, DM, BSZ, DM);
      {
        const float* g = dec_ln + (size_t)((l * 3 + 2) * 2 + 0) * DM;
        ln_kernel<<<BSZ, DM, 0, stream>>>(hB, g, g + DM);
      }
      { float* tmp = hA; hA = hB; hB = tmp; }
    }

    g128(hA, wqFt, nullptr, qt, DM, BSZ, DM);
    pointer_kernel<<<BSZ, 64, 0, stream>>>(qt, K_fin, maskp, idxp, slp,
                                           out_tours, out_scores, t, (t == NN - 1) ? 1 : 0);
  }
}
